// MassMatrixNet_38843684225964
// MI455X (gfx1250) — matrix operand order repaired and checked by reading, not compiled, not run on hardware
//
#include <hip/hip_runtime.h>
#include <hip/hip_bf16.h>
#include <cstdint>

// ---------------------------------------------------------------------------
// MassMatrixNet: M = L L^T where L comes from a 7->128->128->28 softplus MLP.
// Compute-bound: dominated by the 128x128 GEMM -> v_wmma_f32_16x16x32_f16.
// One wave (32 lanes) owns a 16-row batch tile. 8 waves / block = 128 rows.
// Activation path hand-tuned to raw v_exp_f32/v_log_f32 (VALU is the limiter;
// WMMA co-executes with it).
// ---------------------------------------------------------------------------

typedef __attribute__((ext_vector_type(16))) _Float16 v16h;
typedef __attribute__((ext_vector_type(8)))  _Float16 v8h;
typedef __attribute__((ext_vector_type(8)))  float    v8f;
typedef __attribute__((ext_vector_type(2)))  float    v2f;

#define HID     128
#define WT_STR  136   // padded half-stride for transposed weight tiles in LDS
#define H_STR   136   // padded half-stride for h2 staging in LDS

// LDS layout element counts (must match carve order in kernel)
#define WT_HALVES   ((128 + 32) * WT_STR)          // Wt2 (128 rows) + Wt3 (32 rows)
#define F32_CNT     (7 * 128 + 128 + 128 + 32)     // W1, b1, b2, b3(padded)
#define H2_HALVES   (8 * 16 * H_STR)               // per-wave h2 staging
#define S_F32       (8 * 16 * 49)                  // per-wave raw/M scratch
#define SMEM_BYTES  (WT_HALVES * 2 + F32_CNT * 4 + H2_HALVES * 2 + S_F32 * 4)

// Fast, stable softplus: max(x,0) + ln2*log2(1 + 2^(-|x|*log2e)).
// The log argument is always in [1,2] and the exp argument is <= 0, so the
// raw hardware transcendentals need no range fixup -> ~7 VALU, 2 trans.
__device__ __forceinline__ float softplusf(float x) {
    float ax = __builtin_fabsf(x);
    float e  = __builtin_amdgcn_exp2f(-1.44269504088896340f * ax);
    float l  = 0.69314718055994531f * __builtin_amdgcn_logf(1.0f + e);
    return fmaxf(x, 0.0f) + l;
}

__device__ __forceinline__ v16h cat16(v8h lo, v8h hi) {
    return __builtin_shufflevector(lo, hi, 0, 1, 2, 3, 4, 5, 6, 7,
                                           8, 9, 10, 11, 12, 13, 14, 15);
}

// A-matrix fragment (16x32 f16) from row-major [16][H_STR] staging.
// Lane layout (ISA 7.12.2): row m = lane&15; halves e<8 -> K = ks*32+hi*8+e,
// e>=8 -> K = ks*32+hi*8+e+8. Both quads are 16B-contiguous in LDS.
__device__ __forceinline__ v16h load_fragA(const _Float16* base, int m, int ks, int hi) {
    const _Float16* p = base + m * H_STR + ks * 32 + hi * 8;
    v8h lo = *(const v8h*)p;
    v8h hh = *(const v8h*)(p + 16);
    return cat16(lo, hh);
}

// B-matrix fragment (32x16 f16) from transposed weights Wt[n][k] ([*][WT_STR]).
// Lane layout: col n = lane&15; e<8 -> K=ks*32+hi*8+e, e>=8 -> K=ks*32+hi*8+e+8
// (matches A fragment) -> two 16B-contiguous quads per lane.
__device__ __forceinline__ v16h load_fragB(const _Float16* wt, int n, int ks, int hi) {
    const _Float16* p = wt + n * WT_STR + ks * 32 + hi * 8;
    v8h lo = *(const v8h*)p;
    v8h hh = *(const v8h*)(p + 16);
    return cat16(lo, hh);
}

__global__ void __launch_bounds__(256)
massnet_kernel(const float* __restrict__ q,
               const float* __restrict__ W1, const float* __restrict__ b1,
               const float* __restrict__ W2, const float* __restrict__ b2,
               const float* __restrict__ W3, const float* __restrict__ b3,
               float* __restrict__ out, int nrows)
{
    extern __shared__ __align__(16) char smem[];
    _Float16* sWt2 = (_Float16*)smem;                 // [128][WT_STR]  W2^T f16
    _Float16* sWt3 = sWt2 + 128 * WT_STR;             // [32][WT_STR]   W3^T f16 (cols 28..31 = 0)
    float*    sW1  = (float*)(sWt3 + 32 * WT_STR);    // [7][128]
    float*    sB1  = sW1 + 7 * 128;                   // [128]
    float*    sB2  = sB1 + 128;                       // [128]
    float*    sB3  = sB2 + 128;                       // [32] (zero padded)
    _Float16* sH2a = (_Float16*)(sB3 + 32);           // 8 x [16][H_STR] h2 staging
    float*    sSa  = (float*)(sH2a + 8 * 16 * H_STR); // 8 x [16][49] raw/M scratch

    const int t = threadIdx.x;

    // ---- cooperative weight preload (once per block) ----
    for (int i = t; i < 7 * 128; i += 256) sW1[i] = W1[i];
    for (int i = t; i < 128; i += 256) { sB1[i] = b1[i]; sB2[i] = b2[i]; }
    for (int i = t; i < 32; i += 256)  sB3[i] = (i < 28) ? b3[i] : 0.0f;
    for (int i = t; i < 128 * 128; i += 256) {
        int n = i >> 7, k = i & 127;
        sWt2[n * WT_STR + k] = (_Float16)W2[k * 128 + n];
    }
    for (int i = t; i < 32 * 128; i += 256) {
        int n = i >> 7, k = i & 127;
        sWt3[n * WT_STR + k] = (n < 28) ? (_Float16)W3[k * 28 + n] : (_Float16)0.0f;
    }
    __syncthreads();

    const int wave = t >> 5;
    const int lane = t & 31;
    const int m    = lane & 15;   // batch-row within tile (A) / out-col (B,C,D)
    const int hi   = lane >> 4;   // lane half select
    const long tile0 = (long)blockIdx.x * 128 + wave * 16;
    if (tile0 >= nrows) return;   // wave-uniform skip (after the only barrier)

    _Float16* hB = sH2a + wave * 16 * H_STR;
    float*    sS = sSa  + wave * 16 * 49;

    // ---- layer 1: h1 = softplus(q@W1+b1), built DIRECTLY in A-fragment layout.
    // Column pairs (e, e+1) are adjacent -> float2 math (v_pk_fma_f32) and
    // ds_load_b64 for the weight/bias reads.
    long qrow = tile0 + m; if (qrow >= nrows) qrow = nrows - 1;
    const float* qr = q + (size_t)qrow * 7;
    float qv[7];
#pragma unroll
    for (int k = 0; k < 7; ++k) qv[k] = qr[k];

    v16h A[4];
#pragma unroll
    for (int ks = 0; ks < 4; ++ks) {
#pragma unroll
        for (int e = 0; e < 16; e += 2) {
            const int col = ks * 32 + hi * 8 + ((e < 8) ? e : e + 8); // even
            v2f acc = *(const v2f*)(sB1 + col);
#pragma unroll
            for (int k = 0; k < 7; ++k)
                acc += qv[k] * (*(const v2f*)(sW1 + k * 128 + col));
            A[ks][e]     = (_Float16)softplusf(acc.x);
            A[ks][e + 1] = (_Float16)softplusf(acc.y);
        }
    }

    // ---- layer 2: h2 = softplus(h1@W2+b2)  (32 WMMAs per tile) ----
#pragma unroll
    for (int nt = 0; nt < 8; ++nt) {
        const float bias = sB2[nt * 16 + m];
        v8f c;
#pragma unroll
        for (int r = 0; r < 8; ++r) c[r] = bias;
#pragma unroll
        for (int ks = 0; ks < 4; ++ks) {
            v16h B = load_fragB(sWt2, nt * 16 + m, ks, hi);
            c = __builtin_amdgcn_wmma_f32_16x16x32_f16(
                    false, A[ks], false, B, (short)0, c, false, false);
        }
#pragma unroll
        for (int r = 0; r < 8; ++r)   // D: row = r + 8*hi, col = nt*16 + m
            hB[(r + hi * 8) * H_STR + nt * 16 + m] = (_Float16)softplusf(c[r]);
    }
    __asm__ volatile("s_wait_dscnt 0" ::: "memory");  // cross-lane staging fence

    // ---- layer 3: raw = h2@W3+b3 (28 cols padded to 32; 8 WMMAs) ----
    v16h C4[4];
#pragma unroll
    for (int ks = 0; ks < 4; ++ks) C4[ks] = load_fragA(hB, m, ks, hi);
#pragma unroll
    for (int nt = 0; nt < 2; ++nt) {
        const int col = nt * 16 + m;
        const float bias = sB3[col];
        v8f c;
#pragma unroll
        for (int r = 0; r < 8; ++r) c[r] = bias;
#pragma unroll
        for (int ks = 0; ks < 4; ++ks) {
            v16h B = load_fragB(sWt3, col, ks, hi);
            c = __builtin_amdgcn_wmma_f32_16x16x32_f16(
                    false, C4[ks], false, B, (short)0, c, false, false);
        }
#pragma unroll
        for (int r = 0; r < 8; ++r)
            sS[(r + hi * 8) * 49 + col] = c[r];
    }
    __asm__ volatile("s_wait_dscnt 0" ::: "memory");

    // ---- build L, M = L L^T per row (lanes 0..15, one row each) ----
    if (lane < 16) {
        float raw[28];
#pragma unroll
        for (int i = 0; i < 28; ++i) raw[i] = sS[lane * 49 + i];
#pragma unroll
        for (int i = 0; i < 7; ++i) {          // diagonal: softplus + eps
            const int d = i * (i + 1) / 2 + i;
            raw[d] = softplusf(raw[d]) + 1e-4f;
        }
#pragma unroll
        for (int i = 0; i < 7; ++i) {
#pragma unroll
            for (int j = 0; j <= i; ++j) {
                float s = 0.0f;
#pragma unroll
                for (int k = 0; k <= j; ++k)
                    s = __builtin_fmaf(raw[i * (i + 1) / 2 + k],
                                       raw[j * (j + 1) / 2 + k], s);
                sS[lane * 49 + i * 7 + j] = s;
                if (j < i) sS[lane * 49 + j * 7 + i] = s;
            }
        }
    }
    __asm__ volatile("s_wait_dscnt 0" ::: "memory");

    // ---- coalesced store of 16 rows x 49 floats ----
    long nvalid = nrows - tile0; if (nvalid > 16) nvalid = 16;
    const int cnt = (int)nvalid * 49;
    float* og = out + (size_t)tile0 * 49;
    for (int i = lane; i < cnt; i += 32) og[i] = sS[i];
}

extern "C" void kernel_launch(void* const* d_in, const int* in_sizes, int n_in,
                              void* d_out, int out_size, void* d_ws, size_t ws_size,
                              hipStream_t stream) {
    const float* q  = (const float*)d_in[0];
    const float* W1 = (const float*)d_in[1];
    const float* b1 = (const float*)d_in[2];
    const float* W2 = (const float*)d_in[3];
    const float* b2 = (const float*)d_in[4];
    const float* W3 = (const float*)d_in[5];
    const float* b3 = (const float*)d_in[6];
    float* out = (float*)d_out;

    const int nrows  = in_sizes[0] / 7;
    const int blocks = (nrows + 127) / 128;

    massnet_kernel<<<blocks, 256, SMEM_BYTES, stream>>>(
        q, W1, b1, W2, b2, W3, b3, out, nrows);
}
